// BasicAttention_partial_2843268350072
// MI455X (gfx1250) — matrix operand order repaired and checked by reading, not compiled, not run on hardware
//
#include <hip/hip_runtime.h>
#include <hip/hip_bf16.h>
#include <stdint.h>

// ---------------------------------------------------------------------------
// BasicAttention (non-local block) fused for gfx1250 (MI455X).
//   B=4, C=256, H=W=64 -> N=4096, IC=128
// Stage 1: QKV 1x1-conv projections as WMMA GEMMs, f32 -> bf16 activations.
// Stage 2: fused flash-attention, double-buffered K/V tiles staged with
//          GLOBAL_LOAD_ASYNC_TO_LDS (ASYNCcnt) when available, software-
//          pipelined ds_load_b128 fragment feeds into v_wmma.
// ---------------------------------------------------------------------------

typedef __attribute__((ext_vector_type(16))) __bf16 v16bf;
typedef __attribute__((ext_vector_type(8)))  __bf16 v8bf;
typedef __attribute__((ext_vector_type(8)))  float  v8f;
typedef int v4i __attribute__((vector_size(16)));   // matches async-LDS builtin

#define NPIX 4096
#define CDIM 256
#define ICDIM 128

#if defined(__has_builtin)
#if __has_builtin(__builtin_amdgcn_global_load_async_to_lds_b128) && \
    __has_builtin(__builtin_amdgcn_s_wait_asynccnt)
#define USE_ASYNC_LDS 1
#endif
#endif
#ifndef USE_ASYNC_LDS
#define USE_ASYNC_LDS 0
#endif

__device__ __forceinline__ __bf16 f2bf(float f) {
    uint32_t u = __builtin_bit_cast(uint32_t, f);
    uint32_t r = (u + 0x7FFFu + ((u >> 16) & 1u)) >> 16;   // RNE
    return __builtin_bit_cast(__bf16, (uint16_t)r);
}
__device__ __forceinline__ uint32_t pack2bf(float a, float b) {
    return (uint32_t)__builtin_bit_cast(uint16_t, f2bf(a)) |
           ((uint32_t)__builtin_bit_cast(uint16_t, f2bf(b)) << 16);
}

// A fragment (16x32 bf16, row-major tile, ld multiple of 8):
// lane row = lane&15; K runs {hi*8..} and {16+hi*8..} contiguous -> 2x b128.
__device__ __forceinline__ v16bf load_a_frag(const __bf16* p, int ld) {
    const int lane = threadIdx.x & 31;
    const int row  = lane & 15;
    const int hi   = lane >> 4;
    v8bf lo = *(const v8bf*)(p + row * ld + hi * 8);
    v8bf hv = *(const v8bf*)(p + row * ld + 16 + hi * 8);
    return __builtin_shufflevector(lo, hv, 0,1,2,3,4,5,6,7,8,9,10,11,12,13,14,15);
}

// B fragment (32x16 bf16) from a COLUMN-MAJOR tile (K fastest, ld = K-stride,
// multiple of 8): lane col = lane&15; K runs {hi*8..} and {16+hi*8..} -> 2x b128.
__device__ __forceinline__ v16bf load_b_frag_cm(const __bf16* p, int ld) {
    const int lane = threadIdx.x & 31;
    const int col  = lane & 15;
    const int hi   = lane >> 4;
    v8bf lo = *(const v8bf*)(p + col * ld + hi * 8);
    v8bf hv = *(const v8bf*)(p + col * ld + 16 + hi * 8);
    return __builtin_shufflevector(lo, hv, 0,1,2,3,4,5,6,7,8,9,10,11,12,13,14,15);
}

// C/D fragment (16x16 f32): element r -> row r + 8*(lane>=16), col lane&15.

// ---------------------------------------------------------------------------
// Stage 1: out = W[O,C] * x[C,N] (+bias) * premul, stored bf16.
//   TR=1: out[col*O + row]   (transposed store: Q^T [N,IC], K^T [N,IC])
//   TR=0: out[row*N + col]   (natural store:    V   [C,N])
// ---------------------------------------------------------------------------
template <bool TR>
__global__ __launch_bounds__(256)
void na_proj_kernel(const float* __restrict__ Wm, const float* __restrict__ bias,
                    const float* __restrict__ x, __bf16* __restrict__ out,
                    int O, float premul) {
    __shared__ __align__(16) __bf16 sA[128 * 32];   // weights [128 r][32 k], row-major
    __shared__ __align__(16) __bf16 sBt[64 * 32];   // x tile  [64 n][32 k], col-major B

    const int b  = blockIdx.z;
    const int m0 = blockIdx.y * 128;
    const int n0 = blockIdx.x * 64;
    const float* xb  = x   + (size_t)b * CDIM * NPIX;
    __bf16*     outb = out + (size_t)b * (size_t)O * NPIX;

    const int tid  = threadIdx.x;
    const int wave = tid >> 5;
    const int lane = tid & 31;

    v8f acc[4];
#pragma unroll
    for (int f = 0; f < 4; ++f) acc[f] = 0.0f;

    for (int k = 0; k < CDIM; k += 32) {
        uint32_t* sA32 = (uint32_t*)sA;
#pragma unroll
        for (int t = 0; t < 8; ++t) {
            int idx = tid + t * 256;                  // 0..2047
            int r = idx >> 4, kp = idx & 15;
            float2 w2 = ((const float2*)(Wm + (size_t)(m0 + r) * CDIM + k))[kp];
            sA32[r * 16 + kp] = pack2bf(w2.x, w2.y);
        }
        uint32_t* sB32 = (uint32_t*)sBt;
#pragma unroll
        for (int t = 0; t < 4; ++t) {
            int idx = tid + t * 256;                  // 0..1023
            int n = idx & 63, kp = idx >> 6;          // kp in [0,16)
            float a0 = xb[(size_t)(k + 2 * kp)     * NPIX + n0 + n];
            float a1 = xb[(size_t)(k + 2 * kp + 1) * NPIX + n0 + n];
            sB32[n * 16 + kp] = pack2bf(a0, a1);
        }
        __syncthreads();

        v16bf a  = load_a_frag(sA + wave * 16 * 32, 32);
        v16bf bb = load_b_frag_cm(sBt, 32);           // f = 0
#pragma unroll
        for (int f = 0; f < 4; ++f) {
            v16bf curb = bb;
            if (f < 3) bb = load_b_frag_cm(sBt + ((f + 1) * 16) * 32, 32);
            acc[f] = __builtin_amdgcn_wmma_f32_16x16x32_bf16(
                false, a, false, curb, (short)0, acc[f], false, false);
        }
        __syncthreads();
    }

    const int hi = lane >> 4, ln = lane & 15;
#pragma unroll
    for (int f = 0; f < 4; ++f) {
#pragma unroll
        for (int r = 0; r < 8; ++r) {
            int row = m0 + wave * 16 + hi * 8 + r;
            int col = n0 + f * 16 + ln;
            float v = (acc[f][r] + bias[row]) * premul;
            if (TR) outb[(size_t)col * O + row]    = f2bf(v);
            else    outb[(size_t)row * NPIX + col] = f2bf(v);
        }
    }
}

// ---------------------------------------------------------------------------
// Tile staging: K tile [64 j][128 ic] (16 KB) + V tile [256 c][64 j] (32 KB).
// Async DMA to LDS (24 b128 ops/thread, ASYNCcnt) when available, else
// VGPR-staged 128-bit copies.
// ---------------------------------------------------------------------------
__device__ __forceinline__ void stage_tiles(const __bf16* Kb, const __bf16* Vb,
                                            int j0, __bf16* dK, __bf16* dV, int tid) {
#if USE_ASYNC_LDS
    typedef __attribute__((address_space(1))) v4i* gp4;
    typedef __attribute__((address_space(3))) v4i* lp4;
    gp4 gk = (gp4)(Kb + (size_t)j0 * ICDIM);
    lp4 lk = (lp4)dK;
#pragma unroll
    for (int t = 0; t < 8; ++t)
        __builtin_amdgcn_global_load_async_to_lds_b128(gk + tid + t * 128,
                                                       lk + tid + t * 128, 0, 0);
    lp4 lv = (lp4)dV;
#pragma unroll
    for (int t = 0; t < 16; ++t) {
        int idx = tid + t * 128;                      // 0..2047
        int c = idx >> 3, q = idx & 7;
        gp4 gv = (gp4)(Vb + (size_t)c * NPIX + j0);
        __builtin_amdgcn_global_load_async_to_lds_b128(gv + q, lv + idx, 0, 0);
    }
#else
    uint4* dk = (uint4*)dK;
    const uint4* sk = (const uint4*)(Kb + (size_t)j0 * ICDIM);
#pragma unroll
    for (int t = 0; t < 8; ++t) dk[tid + t * 128] = sk[tid + t * 128];
    uint4* dv = (uint4*)dV;
#pragma unroll
    for (int t = 0; t < 16; ++t) {
        int idx = tid + t * 128;
        int c = idx >> 3, q = idx & 7;
        dv[idx] = ((const uint4*)(Vb + (size_t)c * NPIX + j0))[q];
    }
#endif
}

// ---------------------------------------------------------------------------
// Stage 2: fused attention.  Block = 128 threads (4 waves); each wave owns
// 16 query rows of a 64-row query block.  Double-buffered 64-key tiles with
// online softmax; output accumulated across full C=256 in registers.
// ---------------------------------------------------------------------------
__global__ __launch_bounds__(128)
void na_attn_kernel(const __bf16* __restrict__ Qt,  // [B][N][IC], scale baked in
                    const __bf16* __restrict__ Kt,  // [B][N][IC]
                    const __bf16* __restrict__ Vg,  // [B][C][N]
                    const float* __restrict__ x,
                    const float* __restrict__ gamma,
                    float* __restrict__ outp) {
    __shared__ __align__(16) __bf16 sKt[2][64 * ICDIM]; // [j][ic], 2 x 16 KB
    __shared__ __align__(16) __bf16 sVc[2][CDIM * 64];  // [c][j],  2 x 32 KB
    __shared__ __align__(16) __bf16 sP[64 * 64];        // P strips,     8 KB

    const int b  = blockIdx.y;
    const int q0 = blockIdx.x * 64;
    const int tid = threadIdx.x, wave = tid >> 5, lane = tid & 31;
    const int hi = lane >> 4, ln = lane & 15;

    const __bf16* Qb = Qt + (size_t)b * NPIX * ICDIM;
    const __bf16* Kb = Kt + (size_t)b * NPIX * ICDIM;
    const __bf16* Vb = Vg + (size_t)b * CDIM * NPIX;

    // Q fragments for this wave's 16 rows: resident in registers all kernel.
    v16bf qa[4];
#pragma unroll
    for (int kk = 0; kk < 4; ++kk)
        qa[kk] = load_a_frag(Qb + (size_t)(q0 + wave * 16) * ICDIM + kk * 32, ICDIM);

    v8f oacc[16];
#pragma unroll
    for (int t = 0; t < 16; ++t) oacc[t] = 0.0f;
    float m[8], l[8];
#pragma unroll
    for (int r = 0; r < 8; ++r) { m[r] = -1e30f; l[r] = 0.0f; }

    stage_tiles(Kb, Vb, 0, sKt[0], sVc[0], tid);       // prologue: fill buffer 0

    for (int j0 = 0; j0 < NPIX; j0 += 64) {
        const int cur = (j0 >> 6) & 1, nxt = cur ^ 1;
        const bool has_next = (j0 + 64 < NPIX);
        if (has_next) stage_tiles(Kb, Vb, j0 + 64, sKt[nxt], sVc[nxt], tid);
#if USE_ASYNC_LDS
        // async loads complete in order: <=24 outstanding => current tile done
        if (has_next) __builtin_amdgcn_s_wait_asynccnt(24);
        else          __builtin_amdgcn_s_wait_asynccnt(0);
#endif
        __syncthreads();                               // current tile visible

        const __bf16* kTile = sKt[cur];
        const __bf16* vTile = sVc[cur];

        // ---- S = (Q*scale) . K^T : 16 WMMAs, B-frags pipelined 1 deep ----
        v8f s[4];
#pragma unroll
        for (int f = 0; f < 4; ++f) s[f] = 0.0f;
        v16bf bb = load_b_frag_cm(kTile, ICDIM);       // kk=0, f=0
#pragma unroll
        for (int kk = 0; kk < 4; ++kk) {
#pragma unroll
            for (int f = 0; f < 4; ++f) {
                v16bf curb = bb;
                const int nf  = (f + 1) & 3;
                const int nkk = (f == 3) ? kk + 1 : kk;
                if (nkk < 4)
                    bb = load_b_frag_cm(kTile + (nf * 16) * ICDIM + nkk * 32, ICDIM);
                s[f] = __builtin_amdgcn_wmma_f32_16x16x32_bf16(
                    false, qa[kk], false, curb, (short)0, s[f], false, false);
            }
        }

        // ---- online softmax (row stats in C-frag layout) ----
        float mnew[8], corr[8];
#pragma unroll
        for (int r = 0; r < 8; ++r) {
            float rm = s[0][r];
#pragma unroll
            for (int f = 1; f < 4; ++f) rm = fmaxf(rm, s[f][r]);
#pragma unroll
            for (int off = 8; off > 0; off >>= 1)
                rm = fmaxf(rm, __shfl_xor(rm, off, 32));
            mnew[r] = fmaxf(m[r], rm);
            corr[r] = __expf(m[r] - mnew[r]);
        }
#pragma unroll
        for (int r = 0; r < 8; ++r) {
            float rs = 0.0f;
#pragma unroll
            for (int f = 0; f < 4; ++f) {
                float p = __expf(s[f][r] - mnew[r]);
                rs += p;
                sP[(wave * 16 + hi * 8 + r) * 64 + f * 16 + ln] = f2bf(p);
            }
#pragma unroll
            for (int off = 8; off > 0; off >>= 1) rs += __shfl_xor(rs, off, 32);
            l[r] = l[r] * corr[r] + rs;
            m[r] = mnew[r];
        }
#pragma unroll
        for (int t = 0; t < 16; ++t)
#pragma unroll
            for (int r = 0; r < 8; ++r) oacc[t][r] *= corr[r];

        // ---- out += P . V^T : 32 WMMAs, B-frags pipelined 1 deep ----
        v16bf pa = load_a_frag(sP + wave * 16 * 64, 64);        // kk=0
        v16bf bv = load_b_frag_cm(vTile, 64);                   // kk=0, t=0
#pragma unroll
        for (int kk = 0; kk < 2; ++kk) {
#pragma unroll
            for (int t = 0; t < 16; ++t) {
                v16bf curb = bv;
                const int nt  = (t + 1) & 15;
                const int nkk = (t == 15) ? kk + 1 : kk;
                if (nkk < 2)
                    bv = load_b_frag_cm(vTile + (nt * 16) * 64 + nkk * 32, 64);
                oacc[t] = __builtin_amdgcn_wmma_f32_16x16x32_bf16(
                    false, pa, false, curb, (short)0, oacc[t], false, false);
            }
            if (kk == 0) pa = load_a_frag(sP + wave * 16 * 64 + 32, 64);
        }
        __syncthreads();        // all waves done with cur before it is refilled
    }

    // ---- epilogue: gamma * (out/l) + 2x ----
    const float g = gamma[0];
    const float* xb = x    + (size_t)b * CDIM * NPIX;
    float*       ob = outp + (size_t)b * CDIM * NPIX;
#pragma unroll
    for (int r = 0; r < 8; ++r) {
        float inv = 1.0f / l[r];
        int i = q0 + wave * 16 + hi * 8 + r;
#pragma unroll
        for (int t = 0; t < 16; ++t) {
            int c = t * 16 + ln;
            size_t off = (size_t)c * NPIX + i;
            ob[off] = g * (oacc[t][r] * inv) + 2.0f * xb[off];
        }
    }
}

// ---------------------------------------------------------------------------
extern "C" void kernel_launch(void* const* d_in, const int* in_sizes, int n_in,
                              void* d_out, int out_size, void* d_ws, size_t ws_size,
                              hipStream_t stream) {
    const float* x     = (const float*)d_in[0];
    const float* Wq    = (const float*)d_in[1];
    const float* bq    = (const float*)d_in[2];
    const float* Wk    = (const float*)d_in[3];
    const float* bk    = (const float*)d_in[4];
    const float* Wv    = (const float*)d_in[5];
    const float* bv    = (const float*)d_in[6];
    const float* gamma = (const float*)d_in[7];
    float* out = (float*)d_out;

    // Workspace: Qt [B][N][IC], Kt [B][N][IC], V [B][C][N]  (bf16)
    __bf16* qt = (__bf16*)d_ws;
    __bf16* kt = qt + (size_t)4 * NPIX * ICDIM;
    __bf16* vg = kt + (size_t)4 * NPIX * ICDIM;

    const float qscale = 0.08838834764831845f;   // IC^-0.5

    na_proj_kernel<true ><<<dim3(64, 1, 4), 256, 0, stream>>>(Wq, bq, x, qt, ICDIM, qscale);
    na_proj_kernel<true ><<<dim3(64, 1, 4), 256, 0, stream>>>(Wk, bk, x, kt, ICDIM, 1.0f);
    na_proj_kernel<false><<<dim3(64, 2, 4), 256, 0, stream>>>(Wv, bv, x, vg, CDIM, 1.0f);
    na_attn_kernel<<<dim3(64, 4), 128, 0, stream>>>(qt, kt, vg, x, gamma, out);
}
